// LSTM_model_56487409877295
// MI455X (gfx1250) — matrix operand order repaired and checked by reading, not compiled, not run on hardware
//
#include <hip/hip_runtime.h>
#include <hip/hip_bf16.h>

typedef __attribute__((ext_vector_type(16))) __bf16 bf16x16;
typedef __attribute__((ext_vector_type(8)))  __bf16 bf16x8;
typedef __attribute__((ext_vector_type(4)))  __bf16 bf16x4;
typedef __attribute__((ext_vector_type(8)))  float  f32x8;

#define B_DIM 65536
#define DIN   256
#define HDIM  512
#define NGATE 4

#define M_TILE 128
#define N_TILE 64
#define KT     32
#define KROW   40                     // padded LDS row (bf16 elems): 80B stride, conflict-free
#define KSTEPS ((DIN + HDIM) / KT)    // 24
#define KSPLIT (DIN / KT)             // 8: steps [0,8) read x/U, [8,24) read h_old/W

__device__ __forceinline__ bf16x16 cat8(bf16x8 lo, bf16x8 hi) {
  return __builtin_shufflevector(lo, hi, 0,1,2,3,4,5,6,7,8,9,10,11,12,13,14,15);
}
__device__ __forceinline__ float fast_sigmoid(float x) {
  return 1.0f / (1.0f + __expf(-x));
}
__device__ __forceinline__ float fast_tanh(float x) {
  // saturation-safe: e->inf gives 1, e->0 gives -1
  float e = __expf(2.0f * x);
  return 1.0f - 2.0f / (e + 1.0f);
}

__global__ __launch_bounds__(256)
void lstm_wmma_kernel(const float* __restrict__ x,
                      const float* __restrict__ h_old,
                      const float* __restrict__ c_old,
                      const float* __restrict__ U,
                      const float* __restrict__ bU,
                      const float* __restrict__ W,
                      const float* __restrict__ bW,
                      float* __restrict__ out) {
  // A tiles: [M][K] row-major; B tiles: [N][K] (transposed at store) — both
  // fragments then assemble from two aligned 16B ds_load_b128 per lane.
  __shared__ __bf16 sA[2][M_TILE][KROW];
  __shared__ __bf16 sB[2][NGATE][N_TILE][KROW];

  const int tid  = threadIdx.x;
  const int wave = tid >> 5;
  const int lane = tid & 31;
  const int half = lane >> 4;   // wave32 half
  const int lrow = lane & 15;
  const int wm   = wave & 3;    // 4 wave-rows of 32
  const int wn   = wave >> 2;   // 2 wave-cols of 32

  const int n0 = blockIdx.x * N_TILE;   // column block within H (per gate)
  const int m0 = blockIdx.y * M_TILE;   // batch block

  // --- staging work split ---
  // A: 128x32 f32 = 1024 float4; 256 thr -> 4 passes, 8 float4 per row
  const int a_k4  = tid & 7;
  const int a_row = tid >> 3;
  // B: per gate 32x64 f32; 64 thr per gate, 8 passes
  const int b_g  = tid >> 6;
  const int b_t  = tid & 63;
  const int b_n4 = b_t & 15;
  const int b_k  = b_t >> 4;

  float4 aReg[4];
  float4 bReg[8];

  auto load_global = [&](int kk) {
    const float* aSrc; int lda;
    if (kk < KSPLIT) { aSrc = x     + (size_t)m0 * DIN  + kk * KT;            lda = DIN;  }
    else             { aSrc = h_old + (size_t)m0 * HDIM + (kk - KSPLIT) * KT; lda = HDIM; }
    #pragma unroll
    for (int p = 0; p < 4; ++p)
      aReg[p] = *(const float4*)(aSrc + (size_t)(a_row + p * 32) * lda + a_k4 * 4);
    const float* bSrc;
    if (kk < KSPLIT) bSrc = U + ((size_t)b_g * DIN  + (size_t)kk * KT) * HDIM + n0;
    else             bSrc = W + ((size_t)b_g * HDIM + (size_t)(kk - KSPLIT) * KT) * HDIM + n0;
    #pragma unroll
    for (int p = 0; p < 8; ++p)
      bReg[p] = *(const float4*)(bSrc + (size_t)(b_k + p * 4) * HDIM + b_n4 * 4);
  };

  auto store_lds = [&](int buf) {
    #pragma unroll
    for (int p = 0; p < 4; ++p) {
      bf16x4 v = { (__bf16)aReg[p].x, (__bf16)aReg[p].y,
                   (__bf16)aReg[p].z, (__bf16)aReg[p].w };
      *(bf16x4*)&sA[buf][a_row + p * 32][a_k4 * 4] = v;   // 8B aligned store
    }
    #pragma unroll
    for (int p = 0; p < 8; ++p) {
      const int k = b_k + p * 4;
      sB[buf][b_g][b_n4 * 4 + 0][k] = (__bf16)bReg[p].x;  // transpose into [N][K]
      sB[buf][b_g][b_n4 * 4 + 1][k] = (__bf16)bReg[p].y;
      sB[buf][b_g][b_n4 * 4 + 2][k] = (__bf16)bReg[p].z;
      sB[buf][b_g][b_n4 * 4 + 3][k] = (__bf16)bReg[p].w;
    }
  };

  f32x8 acc[NGATE][2][2] = {};   // 128 accumulator VGPRs / lane

  // A frag (ISA 16-bit A 16x32): lanes 0-15 hold K 0..7 & 16..23; lanes 16-31 K 8..15 & 24..31
  const int aE0 = half * 8;
  // B frag (same K order as A): lanes 0-15 K 0..7 & 16..23; lanes 16-31 K 8..15 & 24..31
  const int bE0 = half * 8;

  load_global(0);
  for (int kk = 0; kk < KSTEPS; ++kk) {
    const int buf = kk & 1;
    store_lds(buf);
    __syncthreads();                       // one barrier per K-step suffices (see pipeline)
    if (kk + 1 < KSTEPS) load_global(kk + 1);   // overlap next global fetch with WMMA burst

    bf16x16 afrag[2];
    #pragma unroll
    for (int mi = 0; mi < 2; ++mi) {
      const __bf16* ap = &sA[buf][wm * 32 + mi * 16 + lrow][0];
      afrag[mi] = cat8(*(const bf16x8*)(ap + aE0), *(const bf16x8*)(ap + aE0 + 16));
    }
    #pragma unroll
    for (int g = 0; g < NGATE; ++g) {
      bf16x16 bfrag[2];
      #pragma unroll
      for (int ni = 0; ni < 2; ++ni) {
        const __bf16* bp = &sB[buf][g][wn * 32 + ni * 16 + lrow][0];
        bfrag[ni] = cat8(*(const bf16x8*)(bp + bE0), *(const bf16x8*)(bp + bE0 + 16));
      }
      #pragma unroll
      for (int mi = 0; mi < 2; ++mi)
        #pragma unroll
        for (int ni = 0; ni < 2; ++ni)
          acc[g][mi][ni] = __builtin_amdgcn_wmma_f32_16x16x32_bf16(
              false, afrag[mi], false, bfrag[ni],
              (short)0, acc[g][mi][ni], false, false);
    }
  }

  // ---- fused LSTM epilogue ----
  float* out_h = out;
  float* out_c = out + (size_t)B_DIM * HDIM;
  #pragma unroll
  for (int ni = 0; ni < 2; ++ni) {
    const int c = n0 + wn * 32 + ni * 16 + lrow;   // C/D layout: N = lane&15
    const float bi  = bU[0 * HDIM + c] + bW[0 * HDIM + c];
    const float bff = bU[1 * HDIM + c] + bW[1 * HDIM + c];
    const float bo  = bU[2 * HDIM + c] + bW[2 * HDIM + c];
    const float bc  = bU[3 * HDIM + c] + bW[3 * HDIM + c];
    #pragma unroll
    for (int mi = 0; mi < 2; ++mi) {
      #pragma unroll
      for (int j = 0; j < 8; ++j) {
        const int r = m0 + wm * 32 + mi * 16 + half * 8 + j;  // C/D: M = j (+8 hi half)
        const size_t idx = (size_t)r * HDIM + c;
        const float it = fast_sigmoid(acc[0][mi][ni][j] + bi);
        const float ft = fast_sigmoid(acc[1][mi][ni][j] + bff);
        const float ot = fast_sigmoid(acc[2][mi][ni][j] + bo);
        const float ct = fast_tanh  (acc[3][mi][ni][j] + bc);
        const float cn = it * ct + ft * c_old[idx];
        out_c[idx] = cn;
        out_h[idx] = ot * fast_tanh(cn);
      }
    }
  }
}

extern "C" void kernel_launch(void* const* d_in, const int* in_sizes, int n_in,
                              void* d_out, int out_size, void* d_ws, size_t ws_size,
                              hipStream_t stream) {
  const float* x    = (const float*)d_in[0];
  const float* h    = (const float*)d_in[1];
  const float* c    = (const float*)d_in[2];
  const float* U    = (const float*)d_in[3];
  const float* bU   = (const float*)d_in[4];
  const float* W    = (const float*)d_in[5];
  const float* bW   = (const float*)d_in[6];
  float* out        = (float*)d_out;

  dim3 grid(HDIM / N_TILE, B_DIM / M_TILE);   // (8, 512)
  lstm_wmma_kernel<<<grid, dim3(256), 0, stream>>>(x, h, c, U, bU, W, bW, out);
}
